// Net_50783693308233
// MI455X (gfx1250) — matrix operand order repaired and checked by reading, not compiled, not run on hardware
//
#include <hip/hip_runtime.h>
#include <hip/hip_bf16.h>

// ---------------------------------------------------------------------------
// GCN (5x GCNConv + 2-layer head) for MI455X / gfx1250.
// GEMMs run on v_wmma_f32_16x16x32_bf16 (f32 accumulate); edge aggregation is
// f32 global atomics (layer feature matrices fit the 192MB L2).
// ---------------------------------------------------------------------------

typedef __bf16 v16bf __attribute__((ext_vector_type(16)));
typedef float  v8f   __attribute__((ext_vector_type(8)));

#define MP      50048      // 50000 padded to multiple of 128
#define F_IN    1022
#define KP1     1024       // F_IN padded to multiple of 32
#define H1      1024
#define H2      512
#define UH      75
#define OUTD    11
#define BM      128
#define BN      64
#define KT      32

// ------------------------------ utility kernels -----------------------------

__global__ void zero_f32(float* p, long n) {
    long i = (long)blockIdx.x * blockDim.x + threadIdx.x;
    long stride = (long)gridDim.x * blockDim.x;
    for (; i < n; i += stride) p[i] = 0.0f;
}

__global__ void deg_accum(const long long* ei, float* deg, int E) {
    int e = blockIdx.x * blockDim.x + threadIdx.x;
    if (e < E) {
        int d = (int)ei[(long)E + e];   // dst row
        atomicAdd(&deg[d], 1.0f);
    }
}

__global__ void make_dinv(const float* deg, float* dinv, int n_real, int n_pad) {
    int i = blockIdx.x * blockDim.x + threadIdx.x;
    if (i < n_pad) dinv[i] = (i < n_real) ? rsqrtf(deg[i] + 1.0f) : 0.0f;
}

// Convert fp32 (rows x cols) -> bf16 (rowsP x colsP), zero padding.
__global__ void cvt_pad_bf16(const float* __restrict__ src, __bf16* __restrict__ dst,
                             long rows, long cols, long rowsP, long colsP) {
    long total = rowsP * colsP;
    long i = (long)blockIdx.x * blockDim.x + threadIdx.x;
    long stride = (long)gridDim.x * blockDim.x;
    for (; i < total; i += stride) {
        long r = i / colsP, c = i % colsP;
        float v = (r < rows && c < cols) ? src[r * cols + c] : 0.0f;
        dst[i] = (__bf16)v;
    }
}

// ------------------------------ WMMA GEMM -----------------------------------
// C[M x N] = A[M x K] * B[K x N]; A,B bf16 row-major, C f32 row-major.
// M % 128 == 0, N % 64 == 0, K % 32 == 0.  Block = 256 threads = 8 waves.
// Block tile 128x64; wave w computes rows [16w,16w+16) x 64 cols (4 wmma accs).
__global__ __launch_bounds__(256)
void gemm_bf16_wmma(const __bf16* __restrict__ A, const __bf16* __restrict__ B,
                    float* __restrict__ C, int K, int N) {
    __shared__ __align__(16) __bf16 sA[BM * KT];      // 8 KB, [row][k]
    __shared__ __align__(16) __bf16 sB[BN * KT];      // 4 KB, [n][k] (transposed)

    const int tid  = threadIdx.x;
    const int lane = tid & 31;
    const int w    = tid >> 5;
    const int m    = lane & 15;       // also n for B / C column
    const int hi   = lane >> 4;       // 0 or 1
    const long bm0 = (long)blockIdx.x * BM;
    const int  bn0 = blockIdx.y * BN;

    v8f acc[4];
    const v8f vzero = {0.f, 0.f, 0.f, 0.f, 0.f, 0.f, 0.f, 0.f};
    acc[0] = vzero; acc[1] = vzero; acc[2] = vzero; acc[3] = vzero;

    // A-tile fill: 256 threads x 8 bf16 (16B) = k 0-7 and 16-23 of the 128x32 tile.
    const int arow = tid >> 1;
    const int akc  = (tid & 1) * 16;
    // B-tile fill: 256 threads x 8 bf16, transposed into sB[n][k].
    const int bnn  = tid & 63;
    const int bkq  = (tid >> 6) * 8;

    for (int k0 = 0; k0 < K; k0 += KT) {
        __syncthreads();
        // load A tile (coalesced 16B per thread)
        *(uint4*)&sA[arow * KT + akc] =
            *(const uint4*)&A[(bm0 + arow) * (long)K + k0 + akc];
        // load B tile transposed
        __align__(16) __bf16 tmp[8];
        #pragma unroll
        for (int kk = 0; kk < 8; ++kk)
            tmp[kk] = B[(long)(k0 + bkq + kk) * N + bn0 + bnn];
        *(uint4*)&sB[bnn * KT + bkq] = *(const uint4*)tmp;
        __syncthreads();

        // A fragment: k order used here (same as B, not AMD's table order):
        //   element e -> k = e + 16*hi, row = w*16 + m
        v16bf afrag;
        const __bf16* pa = &sA[(w * 16 + m) * KT + 16 * hi];
        #pragma unroll
        for (int e = 0; e < 8; ++e) { afrag[e] = pa[e]; afrag[e + 8] = pa[8 + e]; }

        #pragma unroll
        for (int j = 0; j < 4; ++j) {
            // B fragment: element e -> k = e + 16*hi, n = m  (contiguous in sB)
            v16bf bfrag;
            const __bf16* pb = &sB[(j * 16 + m) * KT + 16 * hi];
            #pragma unroll
            for (int e = 0; e < 16; ++e) bfrag[e] = pb[e];

            acc[j] = __builtin_amdgcn_wmma_f32_16x16x32_bf16(
                false, afrag, false, bfrag, (short)0, acc[j], false, false);
        }
    }

    // C/D layout: VGPR r -> row = r + 8*hi, col = m
    #pragma unroll
    for (int j = 0; j < 4; ++j) {
        #pragma unroll
        for (int r = 0; r < 8; ++r) {
            long gm = bm0 + w * 16 + r + 8 * hi;
            int  gn = bn0 + j * 16 + m;
            C[gm * (long)N + gn] = acc[j][r];
        }
    }
}

// ------------------------------ aggregation ---------------------------------

// agg[i] = G[i] * dinv[i]^2   (self-loop term; real rows only)
__global__ void init_agg(const float* __restrict__ G, float* __restrict__ agg,
                         const float* __restrict__ dinv, int F) {
    int i = blockIdx.x;
    float d = dinv[i];
    float s = d * d;
    const float* g = G + (long)i * F;
    float* a = agg + (long)i * F;
    for (int f = threadIdx.x; f < F; f += blockDim.x) a[f] = g[f] * s;
}

// agg[dst] += G[src] * dinv[src]*dinv[dst]   via f32 atomics (L2-resident)
__global__ void edge_agg(const float* __restrict__ G, float* __restrict__ agg,
                         const long long* __restrict__ ei,
                         const float* __restrict__ dinv, int E, int F) {
    int e = blockIdx.x;
    int s = (int)ei[e];
    int d = (int)ei[(long)E + e];
    float nrm = dinv[s] * dinv[d];
    const float* g = G + (long)s * F;
    float* a = agg + (long)d * F;
    for (int f = threadIdx.x; f < F; f += blockDim.x)
        atomicAdd(&a[f], g[f] * nrm);
}

// out_bf16 = bf16(relu(agg + b)); pad rows forced to zero
__global__ void finalize_relu_bf16(const float* __restrict__ agg,
                                   const float* __restrict__ bias,
                                   __bf16* __restrict__ out, int F, int n_real) {
    int i = blockIdx.x;
    __bf16* o = out + (long)i * F;
    if (i < n_real) {
        const float* a = agg + (long)i * F;
        for (int f = threadIdx.x; f < F; f += blockDim.x)
            o[f] = (__bf16)fmaxf(a[f] + bias[f], 0.0f);
    } else {
        for (int f = threadIdx.x; f < F; f += blockDim.x)
            o[f] = (__bf16)0.0f;
    }
}

// ------------------------------ head ----------------------------------------
// per-node: u = relu(h @ Uw1 + Ub1) [75]; out = u @ Uw2 + Ub2 [11]
__global__ __launch_bounds__(128)
void head_kernel(const __bf16* __restrict__ h,
                 const float* __restrict__ Uw1, const float* __restrict__ Ub1,
                 const float* __restrict__ Uw2, const float* __restrict__ Ub2,
                 float* __restrict__ out) {
    __shared__ float sh[H2];
    __shared__ float su[UH + 1];
    int i = blockIdx.x;
    for (int f = threadIdx.x; f < H2; f += blockDim.x)
        sh[f] = (float)h[(long)i * H2 + f];
    __syncthreads();
    if (threadIdx.x < UH) {
        float acc = Ub1[threadIdx.x];
        #pragma unroll 8
        for (int k = 0; k < H2; ++k) acc += sh[k] * Uw1[k * UH + threadIdx.x];
        su[threadIdx.x] = fmaxf(acc, 0.0f);
    }
    __syncthreads();
    if (threadIdx.x < OUTD) {
        float acc = Ub2[threadIdx.x];
        #pragma unroll
        for (int k = 0; k < UH; ++k) acc += su[k] * Uw2[k * OUTD + threadIdx.x];
        out[(long)i * OUTD + threadIdx.x] = acc;
    }
}

// ------------------------------ launcher ------------------------------------

static inline size_t align256(size_t x) { return (x + 255) & ~(size_t)255; }

extern "C" void kernel_launch(void* const* d_in, const int* in_sizes, int n_in,
                              void* d_out, int out_size, void* d_ws, size_t ws_size,
                              hipStream_t stream) {
    const float*     x  = (const float*)d_in[0];
    const long long* ei = (const long long*)d_in[1];   // int64 edge_index [2,E]
    const int N = in_sizes[0] / F_IN;
    const int E = in_sizes[1] / 2;

    const float* W[5] = {(const float*)d_in[2], (const float*)d_in[4],
                         (const float*)d_in[6], (const float*)d_in[8],
                         (const float*)d_in[10]};
    const float* Bv[5] = {(const float*)d_in[3], (const float*)d_in[5],
                          (const float*)d_in[7], (const float*)d_in[9],
                          (const float*)d_in[11]};
    const float* Uw1 = (const float*)d_in[12];
    const float* Ub1 = (const float*)d_in[13];
    const float* Uw2 = (const float*)d_in[14];
    const float* Ub2 = (const float*)d_in[15];
    float* out = (float*)d_out;

    // ---- workspace carve-up ----
    char* ws = (char*)d_ws;
    size_t off = 0;
    float* deg  = (float*)(ws + off); off = align256(off + (size_t)MP * 4);
    float* dinv = (float*)(ws + off); off = align256(off + (size_t)MP * 4);
    __bf16* Wbf = (__bf16*)(ws + off); off = align256(off + (size_t)KP1 * H1 * 2);
    __bf16* Abf = (__bf16*)(ws + off); off = align256(off + (size_t)MP * KP1 * 2);
    float* G    = (float*)(ws + off);  off = align256(off + (size_t)MP * H1 * 4);
    float* Agg  = (float*)(ws + off);  off = align256(off + (size_t)MP * H1 * 4);

    // ---- degree + normalization ----
    zero_f32<<<512, 256, 0, stream>>>(deg, (long)MP);
    deg_accum<<<(E + 255) / 256, 256, 0, stream>>>(ei, deg, E);
    make_dinv<<<(MP + 255) / 256, 256, 0, stream>>>(deg, dinv, N, MP);

    // ---- x -> bf16 padded [MP x KP1] ----
    cvt_pad_bf16<<<4096, 256, 0, stream>>>(x, Abf, N, F_IN, MP, KP1);

    // layer dims: (K_real, K_pad, N_out)
    const int Kre[5] = {F_IN, H1, H2, H2, H2};
    const int Kpd[5] = {KP1,  H1, H2, H2, H2};
    const int Nou[5] = {H1,   H2, H2, H2, H2};

    for (int l = 0; l < 5; ++l) {
        const int Kp = Kpd[l], Nf = Nou[l];
        // weight -> bf16 padded [Kp x Nf]
        cvt_pad_bf16<<<4096, 256, 0, stream>>>(W[l], Wbf, Kre[l], Nf, Kp, Nf);
        // G = Abf @ Wbf
        dim3 grid(MP / BM, Nf / BN);
        gemm_bf16_wmma<<<grid, 256, 0, stream>>>(Abf, Wbf, G, Kp, Nf);
        // aggregation: self-loop init + edge scatter-add
        init_agg<<<N, 256, 0, stream>>>(G, Agg, dinv, Nf);
        edge_agg<<<E, 256, 0, stream>>>(G, Agg, ei, dinv, E, Nf);
        // bias + relu -> bf16 activations for next layer (pad rows zeroed)
        finalize_relu_bf16<<<MP, 256, 0, stream>>>(Agg, Bv[l], Abf, Nf, N);
    }

    // ---- head: 512 -> 75 -> 11 ----
    head_kernel<<<N, 128, 0, stream>>>(Abf, Uw1, Ub1, Uw2, Ub2, out);
}
